// SKLinear_31834297598241
// MI455X (gfx1250) — matrix operand order repaired and checked by reading, not compiled, not run on hardware
//
#include <hip/hip_runtime.h>
#include <hip/hip_bf16.h>

typedef __attribute__((ext_vector_type(16))) _Float16 v16h;
typedef __attribute__((ext_vector_type(8)))  _Float16 v8h;
typedef __attribute__((ext_vector_type(8)))  float    v8f;

#define L_DIM  16
#define K_DIM  64
#define D1     16384
#define D2     16384
#define NROWS  1024   // L*K

// ---------------------------------------------------------------------------
// Kernel 1: z[r] = dot(M[r,:], h) for r in [0,2048)
//   rows 0..1023   -> S1s (z1),  rows 1024..2047 -> U2s (z2)
// One block handles 8 rows; float4 streaming loads; LDS tree reduction.
// h (64 KB) stays resident in the 192 MB L2 across all re-reads.
// ---------------------------------------------------------------------------
__global__ void __launch_bounds__(256) gemv_rows_kernel(
    const float* __restrict__ S1s,   // [1024, 16384]
    const float* __restrict__ U2s,   // [1024, 16384]
    const float* __restrict__ h,     // [16384]
    float* __restrict__ z)           // [2048]
{
    __shared__ float red[256];
    const int tid = threadIdx.x;
    const float4* h4 = (const float4*)h;

    for (int sub = 0; sub < 8; ++sub) {
        const int row = blockIdx.x * 8 + sub;           // 0..2047
        const float* M = (row < NROWS)
                           ? (S1s + (size_t)row * D2)
                           : (U2s + (size_t)(row - NROWS) * D2);
        const float4* M4 = (const float4*)M;

        float acc = 0.f;
        for (int i = tid; i < D2 / 4; i += 256) {
            float4 mv = M4[i];
            float4 hv = h4[i];                          // L2-resident
            acc = fmaf(mv.x, hv.x, acc);
            acc = fmaf(mv.y, hv.y, acc);
            acc = fmaf(mv.z, hv.z, acc);
            acc = fmaf(mv.w, hv.w, acc);
        }
        red[tid] = acc;
        __syncthreads();
        for (int s = 128; s > 0; s >>= 1) {
            if (tid < s) red[tid] += red[tid + s];
            __syncthreads();
        }
        if (tid == 0) z[row] = red[0];
        __syncthreads();
    }
}

// ---------------------------------------------------------------------------
// Kernel 2: t1[o] = sum_r U1s[r, o] * z1[r]   via v_wmma_f32_16x16x32_f16.
//   A[m,k] = z1[r0+k]  (identical rows, broadcast)     -> 16x32 f16
//   B[k,n] = U1s[r0+k, o0+n]                           -> 32x16 f16
//   D[m,n] = t1[o0+n] for every m; accumulate over r0 in C (f32).
// U1 entries are +-0.125 -> exact in f16. One wave per 16-output tile.
//
// A-operand K order here (k = slot + 16*half16, same as B) is two contiguous runs
// of 8 halves: [r0 + 16*half16, +8) and [r0 + 16*half16 + 8, +8), so z1 is
// pre-converted to f16 in LDS once per block and A is built from two 16-byte
// DS loads per iteration. B is the mandatory 64 MB streaming traffic.
// ---------------------------------------------------------------------------
__global__ void __launch_bounds__(256) wmma_t1_kernel(
    const float* __restrict__ U1s,   // [1024, 16384]
    const float* __restrict__ z,     // z1 = z[0..1023]
    float* __restrict__ t1)          // [16384]
{
    __shared__ __align__(16) _Float16 z16[NROWS];      // 2 KB
    const int tid    = threadIdx.x;
    const int lane   = tid & 31;
    const int wave   = tid >> 5;
    const int half16 = lane >> 4;         // 0: lanes 0-15, 1: lanes 16-31
    const int n      = lane & 15;
    const int o0     = (blockIdx.x * 8 + wave) * 16;

    for (int i = tid; i < NROWS; i += 256) z16[i] = (_Float16)z[i];
    __syncthreads();

    v8f c = {};
#pragma unroll 4
    for (int r0 = 0; r0 < NROWS; r0 += 32) {
        // A: two contiguous 8-half runs from LDS (16B ds loads)
        v8h a_lo = *(const v8h*)(&z16[r0 + 16 * half16]);
        v8h a_hi = *(const v8h*)(&z16[r0 + 16 * half16 + 8]);
        v16h a;
#pragma unroll
        for (int j = 0; j < 8; ++j) { a[j] = a_lo[j]; a[j + 8] = a_hi[j]; }

        // B: streaming f32 -> f16; K = slot + 16*half16, column o0+n
        v16h b;
#pragma unroll
        for (int hh = 0; hh < 16; ++hh) {
            const int Kb = hh + 16 * half16;
            b[hh] = (_Float16)U1s[(size_t)(r0 + Kb) * D1 + o0 + n];
        }
        // (neg_a, A, neg_b, B, c_mod, C, reuse_a, reuse_b)
        c = __builtin_amdgcn_wmma_f32_16x16x32_f16(false, a, false, b,
                                                   (short)0, c, false, false);
    }
    // Row M=0 of D lives in VGPR0: lanes 0..15 hold columns n=0..15.
    if (lane < 16) t1[o0 + n] = c[0];
}

// ---------------------------------------------------------------------------
// Kernel 3: out[o] = (t1[o] + sum_{l,k} S2s[l,o,k]*z2[l*64+k]) / 32 + b[o]
// z2 staged in LDS (4 KB); each thread streams 16 contiguous 64-float chunks.
// ---------------------------------------------------------------------------
__global__ void __launch_bounds__(256) finalize_kernel(
    const float* __restrict__ S2s,   // [16, 16384, 64]
    const float* __restrict__ z,     // z2 = z[1024..2047]
    const float* __restrict__ t1,    // [16384]
    const float* __restrict__ bvec,  // [16384]
    float* __restrict__ out)         // [16384]
{
    __shared__ float z2s[NROWS];     // 4 KB
    const int tid = threadIdx.x;
    for (int i = tid; i < NROWS; i += 256) z2s[i] = z[NROWS + i];
    __syncthreads();

    const int o = blockIdx.x * 256 + tid;
    float acc = 0.f;
#pragma unroll
    for (int l = 0; l < L_DIM; ++l) {
        const float4* row = (const float4*)(S2s + ((size_t)l * D1 + o) * K_DIM);
        const float4* zz  = (const float4*)(z2s + l * K_DIM);
#pragma unroll
        for (int k4 = 0; k4 < K_DIM / 4; ++k4) {
            float4 sv = row[k4];
            float4 zv = zz[k4];
            acc = fmaf(sv.x, zv.x, acc);
            acc = fmaf(sv.y, zv.y, acc);
            acc = fmaf(sv.z, zv.z, acc);
            acc = fmaf(sv.w, zv.w, acc);
        }
    }
    out[o] = (t1[o] + acc) * (1.0f / 32.0f) + bvec[o];
}

// ---------------------------------------------------------------------------
// Host-side launch: inputs in setup_inputs() order:
//   0:h_in [16384]  1:U1s [16,64,16384]  2:U2s [16,64,16384]
//   3:S1s [16,64,16384]  4:S2s [16,16384,64]  5:b [16384]
// Workspace: z (2048 f32) then t1 (16384 f32) = 73,728 bytes.
// ---------------------------------------------------------------------------
extern "C" void kernel_launch(void* const* d_in, const int* in_sizes, int n_in,
                              void* d_out, int out_size, void* d_ws, size_t ws_size,
                              hipStream_t stream) {
    const float* h_in = (const float*)d_in[0];
    const float* U1s  = (const float*)d_in[1];
    const float* U2s  = (const float*)d_in[2];
    const float* S1s  = (const float*)d_in[3];
    const float* S2s  = (const float*)d_in[4];
    const float* bv   = (const float*)d_in[5];
    float* out = (float*)d_out;

    float* z  = (float*)d_ws;        // 2048 floats
    float* t1 = z + 2048;            // 16384 floats

    // Phase 1: z1 = S1 @ h, z2 = U2 @ h   (streams 128 MB)
    gemv_rows_kernel<<<256, 256, 0, stream>>>(S1s, U2s, h_in, z);
    // Phase 2: t1 = U1^T z1 via WMMA      (streams 64 MB)
    wmma_t1_kernel<<<128, 256, 0, stream>>>(U1s, z, t1);
    // Phase 3: t2 + combine + bias        (streams 64 MB)
    finalize_kernel<<<64, 256, 0, stream>>>(S2s, z, t1, bv, out);
}
